// MultiLoRALinear_19361712571085
// MI455X (gfx1250) — compile-verified, matrix operand order checked by reading, not run on hardware
//
#include <hip/hip_runtime.h>

// MultiLoRALinear for MI455X (gfx1250, wave32, WMMA).
//
// Exact rewrite: since ALL adapters are applied and summed,
//   out = x @ (W + SCALE * sum_t Bm[t]@A[t])^T + b
// Kernel 1 builds W_eff in fp32 (accurate) -> f16 in d_ws (2 MB).
// Kernel 2 is a single 32768x1024x1024 GEMM using v_wmma_f32_16x16x32_f16.
// Pipeline: x (HBM stream) register-prefetched one K-step ahead; W_eff
// (L2-resident) loaded at iteration top; double-buffered LDS with a single
// LDS-only split barrier per 64-wide K step (s_wait_dscnt + s_barrier_*),
// so the barrier does NOT drain the outstanding x prefetch loads.

#define D_IN   1024
#define D_OUT  1024
#define M_TOT  (4 * 8192)
#define LORA_T 16
#define RANK_R 16
#define LSCALE 2.0f   // ALPHA / RANK = 32/16

typedef __attribute__((ext_vector_type(4)))  float    v4f;
typedef __attribute__((ext_vector_type(4)))  _Float16 v4h;
typedef __attribute__((ext_vector_type(8)))  _Float16 v8h;
typedef __attribute__((ext_vector_type(16))) _Float16 v16h;
typedef __attribute__((ext_vector_type(8)))  float    v8f;

// LDS-only workgroup barrier: waits only DScnt (LDS visibility within the
// WGP), then split signal/wait on the workgroup barrier (-1). Unlike
// __syncthreads(), this does not fence global memory, so outstanding HBM
// prefetch loads (loadcnt) remain in flight across the barrier.
__device__ __forceinline__ void lds_barrier() {
    asm volatile("s_wait_dscnt 0x0\n\t"
                 "s_barrier_signal -1\n\t"
                 "s_barrier_wait -1" ::: "memory");
}

// ---------------------------------------------------------------------------
// Kernel 1: W_eff[o][i] = W[o][i] + LSCALE * sum_{t,r} Bm[t][o][r] * A[t][r][i]
// fp32 accumulation, f16 output. 64x64 tile per block, 256 threads.
// ---------------------------------------------------------------------------
__global__ __launch_bounds__(256)
void MultiLoRALinear_build_weff(const float* __restrict__ W,
                                const float* __restrict__ A,
                                const float* __restrict__ Bm,
                                unsigned short* __restrict__ weff /* f16 bits */) {
    __shared__ float BmS[64][17];   // [o][r], padded
    __shared__ float AS[16][68];    // [r][i], padded

    const int tid = threadIdx.x;
    const int i0  = blockIdx.x * 64;
    const int o0  = blockIdx.y * 64;
    const int tx  = tid & 15;   // i group (4 consecutive i)
    const int ty  = tid >> 4;   // o group (4 consecutive o)

    float acc[4][4];
#pragma unroll
    for (int j = 0; j < 4; ++j)
#pragma unroll
        for (int k = 0; k < 4; ++k) acc[j][k] = 0.0f;

    for (int t = 0; t < LORA_T; ++t) {
        __syncthreads();
        {
            // Bm[t][o0+o][r0..r0+3] : 64x16 floats, 4 per thread
            const int o  = tid >> 2;
            const int r0 = (tid & 3) * 4;
            const float4 bv = *(const float4*)(Bm + ((size_t)(t * 1024 + o0 + o)) * 16 + r0);
            BmS[o][r0 + 0] = bv.x; BmS[o][r0 + 1] = bv.y;
            BmS[o][r0 + 2] = bv.z; BmS[o][r0 + 3] = bv.w;
            // A[t][r][i0+iq..iq+3] : 16x64 floats, 4 per thread
            const int r  = tid >> 4;
            const int iq = (tid & 15) * 4;
            const float4 av = *(const float4*)(A + ((size_t)(t * 16 + r)) * 1024 + i0 + iq);
            AS[r][iq + 0] = av.x; AS[r][iq + 1] = av.y;
            AS[r][iq + 2] = av.z; AS[r][iq + 3] = av.w;
        }
        __syncthreads();
#pragma unroll
        for (int r = 0; r < RANK_R; ++r) {
            float av[4], bv[4];
#pragma unroll
            for (int k = 0; k < 4; ++k) av[k] = AS[r][tx * 4 + k];
#pragma unroll
            for (int j = 0; j < 4; ++j) bv[j] = BmS[ty * 4 + j][r];
#pragma unroll
            for (int j = 0; j < 4; ++j)
#pragma unroll
                for (int k = 0; k < 4; ++k) acc[j][k] += bv[j] * av[k];
        }
    }

#pragma unroll
    for (int j = 0; j < 4; ++j) {
        const int o = o0 + ty * 4 + j;
        const v4f wrow = *(const v4f*)(W + (size_t)o * 1024 + i0 + tx * 4);
        v4f res;
#pragma unroll
        for (int k = 0; k < 4; ++k) res[k] = wrow[k] + LSCALE * acc[j][k];
        const v4h h = __builtin_convertvector(res, v4h);
        *(v4h*)(weff + (size_t)o * 1024 + i0 + tx * 4) = h;
    }
}

// ---------------------------------------------------------------------------
// Kernel 2: out[m][n] = sum_k x[m][k] * weff[n][k] + b[n]
// 128x128 block tile, BK=64, 8 waves (4x2), each wave 32x64 (2x4 WMMA frags).
// x register-prefetched; W loaded per-iteration (L2-resident);
// double-buffered LDS, one LDS-only barrier per K step.
// ---------------------------------------------------------------------------
#define BK   64
#define NKT  (D_IN / BK)   // 16 K steps
#define LDT  72            // LDS row pitch in halves (64 + 8 pad -> 144B, conflict-free)
#define TSZ  (128 * LDT)   // halves per tile buffer

__global__ __launch_bounds__(256)
__attribute__((amdgpu_waves_per_eu(1)))
void MultiLoRALinear_gemm(const float* __restrict__ x,
                          const unsigned short* __restrict__ weff,
                          const float* __restrict__ bias,
                          float* __restrict__ out) {
    __shared__ _Float16 XT[2 * TSZ];  // x tile     [m][k], double buffered
    __shared__ _Float16 WT[2 * TSZ];  // weff tile  [n][k], double buffered

    const int tid  = threadIdx.x;
    const int n0   = blockIdx.x * 128;
    const int m0   = blockIdx.y * 128;
    const int wave = tid >> 5;
    const int lane = tid & 31;
    const int wm   = wave & 3;    // 0..3 -> 32-row strip
    const int wn   = wave >> 2;   // 0..1 -> 64-col strip
    const int lh   = lane >> 4;   // lane half
    const int lm   = lane & 15;

    const v8f vzero = {0.f, 0.f, 0.f, 0.f, 0.f, 0.f, 0.f, 0.f};
    v8f acc[2][4];
#pragma unroll
    for (int mi = 0; mi < 2; ++mi)
#pragma unroll
        for (int ni = 0; ni < 4; ++ni) acc[mi][ni] = vzero;

    // ---- prologue: global-load x tile 0 into registers ----
    v4f xr[8];
#pragma unroll
    for (int j = 0; j < 8; ++j) {
        const int c = tid + 256 * j, row = c >> 4, col4 = c & 15;
        xr[j] = *(const v4f*)(x + (size_t)(m0 + row) * D_IN + col4 * 4);
    }

    int buf = 0;
    for (int kt = 0; kt < NKT; ++kt) {
        const int lb = buf * TSZ;
        const int k0 = kt * BK;

        // ---- W tile: issue L2 loads first (in-order VMEM: the ds_store's
        //      wait on these will not wait for the x prefetch issued later) ----
        int4 wtmp[4];
#pragma unroll
        for (int j = 0; j < 4; ++j) {
            const int c = tid + 256 * j, row = c >> 3, q = c & 7;
            wtmp[j] = *(const int4*)(weff + (size_t)(n0 + row) * D_IN + k0 + q * 8);
        }
        // ---- stage x registers -> LDS[buf] (fp32->f16 convert) ----
#pragma unroll
        for (int j = 0; j < 8; ++j) {
            const int c = tid + 256 * j, row = c >> 4, col4 = c & 15;
            const v4h hv = __builtin_convertvector(xr[j], v4h);
            *(v4h*)&XT[lb + row * LDT + col4 * 4] = hv;
        }
        // ---- stage W -> LDS[buf] ----
#pragma unroll
        for (int j = 0; j < 4; ++j) {
            const int c = tid + 256 * j, row = c >> 3, q = c & 7;
            *(int4*)&WT[lb + row * LDT + q * 8] = wtmp[j];
        }
        // ---- prefetch next x tile into registers (overlaps barrier+compute;
        //      the LDS-only barrier below does NOT drain these loads) ----
        if (kt != NKT - 1) {
            const int kn = k0 + BK;
#pragma unroll
            for (int j = 0; j < 8; ++j) {
                const int c = tid + 256 * j, row = c >> 4, col4 = c & 15;
                xr[j] = *(const v4f*)(x + (size_t)(m0 + row) * D_IN + kn + col4 * 4);
            }
        }
        lds_barrier();

        // ---- compute: 2 K-steps of 32, 8 WMMAs each ----
#pragma unroll
        for (int ks = 0; ks < 2; ++ks) {
            // ISA 16-bit A layout: lane half lh -> K base 8*lh for halves [0..7],
            // +16 for halves [8..15].
            v16h fa[2], fb[4];
#pragma unroll
            for (int mi = 0; mi < 2; ++mi) {
                const int off = lb + (wm * 32 + mi * 16 + lm) * LDT + ks * 32 + lh * 8;
                const v8h lo = *(const v8h*)&XT[off];
                const v8h hi = *(const v8h*)&XT[off + 16];
                fa[mi] = __builtin_shufflevector(lo, hi,
                         0,1,2,3,4,5,6,7,8,9,10,11,12,13,14,15);
            }
#pragma unroll
            for (int ni = 0; ni < 4; ++ni) {
                const int off = lb + (wn * 64 + ni * 16 + lm) * LDT + ks * 32 + lh * 8;
                const v8h lo = *(const v8h*)&WT[off];
                const v8h hi = *(const v8h*)&WT[off + 16];
                fb[ni] = __builtin_shufflevector(lo, hi,
                         0,1,2,3,4,5,6,7,8,9,10,11,12,13,14,15);
            }
#pragma unroll
            for (int mi = 0; mi < 2; ++mi)
#pragma unroll
                for (int ni = 0; ni < 4; ++ni)
                    acc[mi][ni] = __builtin_amdgcn_wmma_f32_16x16x32_f16(
                        /*neg_a=*/false, fa[mi], /*neg_b=*/false, fb[ni],
                        /*c_mod=*/(short)0, acc[mi][ni],
                        /*reuse_a=*/false, /*reuse_b=*/false);
        }
        buf ^= 1;
    }

    // Epilogue: C/D layout -> element (VGPR v, lane l): m = v + 8*(l/16), n = l%16.
#pragma unroll
    for (int mi = 0; mi < 2; ++mi) {
#pragma unroll
        for (int ni = 0; ni < 4; ++ni) {
            const int col  = n0 + wn * 64 + ni * 16 + lm;
            const float bv = bias[col];
            const int mrow = m0 + wm * 32 + mi * 16 + lh * 8;
#pragma unroll
            for (int v = 0; v < 8; ++v)
                out[(size_t)(mrow + v) * D_OUT + col] = acc[mi][ni][v] + bv;
        }
    }
}

extern "C" void kernel_launch(void* const* d_in, const int* in_sizes, int n_in,
                              void* d_out, int out_size, void* d_ws, size_t ws_size,
                              hipStream_t stream) {
    const float* x    = (const float*)d_in[0];
    // d_in[1] = weights: dead argument in the reference forward pass.
    const float* W    = (const float*)d_in[2];
    const float* bias = (const float*)d_in[3];
    const float* A    = (const float*)d_in[4];
    const float* Bm   = (const float*)d_in[5];
    float* out = (float*)d_out;

    unsigned short* weff = (unsigned short*)d_ws;  // 1024*1024 f16 = 2 MB

    // Build effective weight (fp32 accumulate, f16 store).
    MultiLoRALinear_build_weff<<<dim3(D_IN / 64, D_OUT / 64), 256, 0, stream>>>(W, A, Bm, weff);

    // Fused GEMM: out = x @ W_eff^T + b.  Grid: 8 N-tiles x 256 M-tiles.
    MultiLoRALinear_gemm<<<dim3(D_OUT / 128, M_TOT / 128), 256, 0, stream>>>(x, weff, bias, out);
}
